// AdjointODEBlock_2791728742460
// MI455X (gfx1250) — matrix operand order repaired and checked by reading, not compiled, not run on hardware
//
#include <hip/hip_runtime.h>
#include <stdint.h>

// ---------------------------------------------------------------------------
// CDNA5 (gfx1250) RK4 neural-ODE: 40 fused bf16-WMMA GEMM pairs.
// B=16384, D=1024, H=2048. wave32, v_wmma_f32_16x16x32_bf16,
// double-buffered GLOBAL_LOAD_ASYNC_TO_LDS staging (ASYNCcnt pipeline).
// ---------------------------------------------------------------------------

typedef __attribute__((ext_vector_type(16))) __bf16 v16bf;
typedef __attribute__((ext_vector_type(8)))  float  v8f;
typedef int v4i __attribute__((vector_size(16)));            // matches builtin proto
typedef __attribute__((address_space(1))) v4i* gas_v4i_ptr;  // global int4*
typedef __attribute__((address_space(3))) v4i* las_v4i_ptr;  // LDS int4*

union FragBF16 {            // 16 bf16 values = 8 VGPRs, loaded as two 16B chunks
    uint4 q[2];
    v16bf v;
};

static __device__ __forceinline__ uint16_t f32_to_bf16(float f) {
    uint32_t u = __builtin_bit_cast(uint32_t, f);
    uint32_t r = u + 0x7FFFu + ((u >> 16) & 1u);   // round-to-nearest-even
    return (uint16_t)(r >> 16);
}

// Tile geometry: block = 256 threads = 8 wave32s in a 4x2 grid.
// Block tile 256x128, BK=32; each wave owns 64x64 = 4x4 WMMA accumulators.
#define BM 256
#define BN 128
#define LDSTRIDE 40   // 32 + 8 halves pad (80B rows, 16B aligned)

// ---------------------------------------------------------------------------
// Async global->LDS staging (6 x b128 per thread per K-step: 4 A + 2 B).
// ---------------------------------------------------------------------------
#if __has_builtin(__builtin_amdgcn_global_load_async_to_lds_b128)
#define ASYNC_BUILTIN 1
#else
#define ASYNC_BUILTIN 0
#endif

static __device__ __forceinline__ void async_copy_b128(const uint16_t* g, uint16_t* l) {
#if ASYNC_BUILTIN
    __builtin_amdgcn_global_load_async_to_lds_b128(
        (gas_v4i_ptr)(uint16_t*)g, (las_v4i_ptr)l, 0, 0);
#else
    const uint64_t ga = (uint64_t)(uintptr_t)g;
    const uint32_t la = (uint32_t)(uintptr_t)(__attribute__((address_space(3))) void*)l;
    asm volatile("global_load_async_to_lds_b128 %0, %1, off"
                 :: "v"(la), "v"(ga) : "memory");
#endif
}

static __device__ __forceinline__ void async_wait_prev() {   // <= 6 outstanding
#if __has_builtin(__builtin_amdgcn_s_wait_asynccnt)
    __builtin_amdgcn_s_wait_asynccnt(6);
#else
    asm volatile("s_wait_asynccnt 0x6" ::: "memory");
#endif
}
static __device__ __forceinline__ void async_wait_all() {
#if __has_builtin(__builtin_amdgcn_s_wait_asynccnt)
    __builtin_amdgcn_s_wait_asynccnt(0);
#else
    asm volatile("s_wait_asynccnt 0x0" ::: "memory");
#endif
}

// Stage one 256x32 A tile and one 128x32 B tile (bf16) into LDS buffers.
static __device__ __forceinline__ void stage_tiles(
    const uint16_t* __restrict__ A, const uint16_t* __restrict__ B,
    int K, int m0, int n0, int k0, uint16_t* sAbuf, uint16_t* sBbuf, int tid)
{
#pragma unroll
    for (int t = 0; t < 4; ++t) {                      // 1024 A chunks of 8 halves
        const int c = tid + t * 256;
        const int row = c >> 2, kc = (c & 3) << 3;
        async_copy_b128(A + (size_t)(m0 + row) * K + k0 + kc,
                        sAbuf + row * LDSTRIDE + kc);
    }
#pragma unroll
    for (int t = 0; t < 2; ++t) {                      // 512 B chunks
        const int c = tid + t * 256;
        const int row = c >> 2, kc = (c & 3) << 3;
        async_copy_b128(B + (size_t)(n0 + row) * K + k0 + kc,
                        sBbuf + row * LDSTRIDE + kc);
    }
}

// ---------------------------------------------------------------------------
// Core: 256x128 block tile, double-buffered async staging, 16 WMMA per wave
// per K-step. A: row-major MxK bf16. B: N-major NxK bf16 (W^T), so both
// fragments are two aligned 16B LDS reads per ISA 7.12.2 layouts.
// ---------------------------------------------------------------------------
static __device__ __forceinline__ void gemm_tile_core(
    const uint16_t* __restrict__ A, const uint16_t* __restrict__ B,
    int K, int m0, int n0, uint16_t* sA, uint16_t* sB, v8f acc[4][4])
{
    const int tid  = threadIdx.x;
    const int lane = tid & 31;
    const int wave = tid >> 5;
    const int wm   = wave >> 1;   // 0..3 : 64-row band
    const int wn   = wave & 1;    // 0..1 : 64-col band
    const int hlf  = lane >> 4;   // lane half (ISA fragment layout)
    const int lm   = lane & 15;

    const v8f vzero = {0.f, 0.f, 0.f, 0.f, 0.f, 0.f, 0.f, 0.f};
#pragma unroll
    for (int i = 0; i < 4; ++i)
#pragma unroll
        for (int j = 0; j < 4; ++j) acc[i][j] = vzero;

    const int nk = K >> 5;
    stage_tiles(A, B, K, m0, n0, 0, sA, sB, tid);      // prologue: group 0

    for (int ks = 0; ks < nk; ++ks) {
        const int cur = ks & 1;
        uint16_t* bufA = sA + cur * (BM * LDSTRIDE);
        uint16_t* bufB = sB + cur * (BN * LDSTRIDE);

        if (ks + 1 < nk) {   // overlap next tile's copies with this tile's WMMA
            stage_tiles(A, B, K, m0, n0, (ks + 1) << 5,
                        sA + (cur ^ 1) * (BM * LDSTRIDE),
                        sB + (cur ^ 1) * (BN * LDSTRIDE), tid);
            async_wait_prev();   // our group ks done (in-order completion)
        } else {
            async_wait_all();
        }
        __syncthreads();         // everyone's group ks visible in LDS

        FragBF16 a[4], b[4];
#pragma unroll
        for (int i = 0; i < 4; ++i) {
            const uint16_t* p = bufA + (wm * 64 + i * 16 + lm) * LDSTRIDE;
            a[i].q[0] = *(const uint4*)(p + hlf * 8);
            a[i].q[1] = *(const uint4*)(p + 16 + hlf * 8);
        }
#pragma unroll
        for (int j = 0; j < 4; ++j) {
            const uint16_t* p = bufB + (wn * 64 + j * 16 + lm) * LDSTRIDE;
            b[j].q[0] = *(const uint4*)(p + hlf * 8);
            b[j].q[1] = *(const uint4*)(p + 16 + hlf * 8);
        }

#pragma unroll
        for (int i = 0; i < 4; ++i)
#pragma unroll
            for (int j = 0; j < 4; ++j)
                acc[i][j] = __builtin_amdgcn_wmma_f32_16x16x32_bf16(
                    false, a[i].v, false, b[j].v,
                    (short)0, acc[i][j], false, false);

        __syncthreads();         // protect buf[cur] before it is re-staged
    }
}

// ---------------------------------------------------------------------------
// GEMM1: act = tanh(X @ W1 + b1)  -> bf16 (16384 x 2048)
// ---------------------------------------------------------------------------
__global__ __launch_bounds__(256) void gemm1_tanh_kernel(
    const uint16_t* __restrict__ X, const uint16_t* __restrict__ W1T,
    const float* __restrict__ b1, uint16_t* __restrict__ Y, int K, int N)
{
    __shared__ __align__(16) uint16_t sA[2][BM * LDSTRIDE];
    __shared__ __align__(16) uint16_t sB[2][BN * LDSTRIDE];
    v8f acc[4][4];
    const int m0 = blockIdx.x * BM, n0 = blockIdx.y * BN;
    gemm_tile_core(X, W1T, K, m0, n0, &sA[0][0], &sB[0][0], acc);

    const int lane = threadIdx.x & 31, wave = threadIdx.x >> 5;
    const int wm = wave >> 1, wn = wave & 1, hlf = lane >> 4, lm = lane & 15;
#pragma unroll
    for (int i = 0; i < 4; ++i)
#pragma unroll
        for (int j = 0; j < 4; ++j) {
            const int col = n0 + wn * 64 + j * 16 + lm;
            const float bb = b1[col];
#pragma unroll
            for (int v = 0; v < 8; ++v) {
                const int row = m0 + wm * 64 + i * 16 + hlf * 8 + v;
                Y[(size_t)row * N + col] = f32_to_bf16(tanhf(acc[i][j][v] + bb));
            }
        }
}

// ---------------------------------------------------------------------------
// GEMM2 + fused RK4 epilogue:
//   k      = act @ W2 + b2
//   accout = accin + g * k          (fp32 RK4 accumulator)
//   xnext  = bf16(xbase + c * k)    (next stage's GEMM1 input)
// ---------------------------------------------------------------------------
__global__ __launch_bounds__(256) void gemm2_rk4_kernel(
    const uint16_t* __restrict__ Abf, const uint16_t* __restrict__ W2T,
    const float* __restrict__ b2,
    const float* __restrict__ accin, float* __restrict__ accout,
    const float* __restrict__ xbase, uint16_t* __restrict__ xnext,
    float g, float c, int K, int N)
{
    __shared__ __align__(16) uint16_t sA[2][BM * LDSTRIDE];
    __shared__ __align__(16) uint16_t sB[2][BN * LDSTRIDE];
    v8f acc[4][4];
    const int m0 = blockIdx.x * BM, n0 = blockIdx.y * BN;
    gemm_tile_core(Abf, W2T, K, m0, n0, &sA[0][0], &sB[0][0], acc);

    const int lane = threadIdx.x & 31, wave = threadIdx.x >> 5;
    const int wm = wave >> 1, wn = wave & 1, hlf = lane >> 4, lm = lane & 15;
#pragma unroll
    for (int i = 0; i < 4; ++i)
#pragma unroll
        for (int j = 0; j < 4; ++j) {
            const int col = n0 + wn * 64 + j * 16 + lm;
            const float bb = b2[col];
#pragma unroll
            for (int v = 0; v < 8; ++v) {
                const int row = m0 + wm * 64 + i * 16 + hlf * 8 + v;
                const size_t idx = (size_t)row * N + col;
                const float kv = acc[i][j][v] + bb;
                const float av = accin[idx];      // read BEFORE the aliased store
                const float xv = xbase[idx];
                accout[idx] = av + g * kv;
                xnext[idx]  = f32_to_bf16(xv + c * kv);
            }
        }
}

// ---------------------------------------------------------------------------
// Weight transpose+convert:  W (R x C, fp32, row-major) -> WT (C x R, bf16)
// ---------------------------------------------------------------------------
__global__ void transpose_bf16_kernel(const float* __restrict__ W,
                                      uint16_t* __restrict__ WT, int R, int C)
{
    const int k = blockIdx.y * 256 + threadIdx.x;   // 0..R-1
    const int n = blockIdx.x;                       // 0..C-1
    if (k < R) WT[(size_t)n * R + k] = f32_to_bf16(W[(size_t)k * C + n]);
}

// hc0 = h (fp32 working state in d_out), xbf = bf16(h) for the first GEMM1
__global__ void init_state_kernel(const float* __restrict__ h,
                                  float* __restrict__ hc,
                                  uint16_t* __restrict__ xbf, size_t n)
{
    const size_t i = (size_t)blockIdx.x * 256 + threadIdx.x;
    if (i < n) {
        const float v = h[i];
        hc[i]  = v;
        xbf[i] = f32_to_bf16(v);
    }
}

// ---------------------------------------------------------------------------
extern "C" void kernel_launch(void* const* d_in, const int* in_sizes, int n_in,
                              void* d_out, int out_size, void* d_ws, size_t ws_size,
                              hipStream_t stream)
{
    (void)in_sizes; (void)n_in; (void)out_size; (void)ws_size;
    const int Bn = 16384, D = 1024, H = 2048;

    const float* h  = (const float*)d_in[0];
    const float* W1 = (const float*)d_in[1];
    const float* b1 = (const float*)d_in[2];
    const float* W2 = (const float*)d_in[3];
    const float* b2 = (const float*)d_in[4];
    float* out = (float*)d_out;

    char* ws = (char*)d_ws;
    size_t off = 0;
    float*    accws = (float*)(ws + off);    off += (size_t)Bn * D * 4;   // 64 MB
    uint16_t* xbf   = (uint16_t*)(ws + off); off += (size_t)Bn * D * 2;   // 32 MB
    uint16_t* act   = (uint16_t*)(ws + off); off += (size_t)Bn * H * 2;   // 64 MB
    uint16_t* w1t   = (uint16_t*)(ws + off); off += (size_t)D * H * 2;    //  4 MB
    uint16_t* w2t   = (uint16_t*)(ws + off);                              //  4 MB

    const dim3 blk(256);

    // Weight prep: W1T = W1^T (H x D), W2T = W2^T (D x H), both bf16
    transpose_bf16_kernel<<<dim3(H, D / 256), blk, 0, stream>>>(W1, w1t, D, H);
    transpose_bf16_kernel<<<dim3(D, H / 256), blk, 0, stream>>>(W2, w2t, H, D);

    // hc0 = h in d_out; xbf = bf16(h)
    const size_t nBD = (size_t)Bn * D;
    init_state_kernel<<<dim3((unsigned)(nBD / 256)), blk, 0, stream>>>(h, out, xbf, nBD);

    const float dt = 0.1f, hdt = 0.05f, g16 = dt / 6.0f, g13 = dt / 3.0f;
    const dim3 grid1(Bn / BM, H / BN);   // 64 x 16
    const dim3 grid2(Bn / BM, D / BN);   // 64 x 8

    for (int s = 0; s < 10; ++s) {
        float* hc  = (s & 1) ? accws : out;   // ping-pong: final acc lands in d_out
        float* acc = (s & 1) ? out   : accws;

        // stage 1: k1; acc = hc + (dt/6)k1; x = bf16(hc + (dt/2)k1)
        gemm1_tanh_kernel<<<grid1, blk, 0, stream>>>(xbf, w1t, b1, act, D, H);
        gemm2_rk4_kernel <<<grid2, blk, 0, stream>>>(act, w2t, b2, hc, acc, hc, xbf, g16, hdt, H, D);
        // stage 2: k2; acc += (dt/3)k2; x = bf16(hc + (dt/2)k2)
        gemm1_tanh_kernel<<<grid1, blk, 0, stream>>>(xbf, w1t, b1, act, D, H);
        gemm2_rk4_kernel <<<grid2, blk, 0, stream>>>(act, w2t, b2, acc, acc, hc, xbf, g13, hdt, H, D);
        // stage 3: k3; acc += (dt/3)k3; x = bf16(hc + dt*k3)
        gemm1_tanh_kernel<<<grid1, blk, 0, stream>>>(xbf, w1t, b1, act, D, H);
        gemm2_rk4_kernel <<<grid2, blk, 0, stream>>>(act, w2t, b2, acc, acc, hc, xbf, g13, dt, H, D);
        // stage 4: k4; acc += (dt/6)k4 = h_{n+1}; x = bf16(h_{n+1}) for next step
        gemm1_tanh_kernel<<<grid1, blk, 0, stream>>>(xbf, w1t, b1, act, D, H);
        gemm2_rk4_kernel <<<grid2, blk, 0, stream>>>(act, w2t, b2, acc, acc, acc, xbf, g16, g16, H, D);
    }
}
